// Nmp_4157528342836
// MI455X (gfx1250) — matrix operand order repaired and checked by reading, not compiled, not run on hardware
//
#include <hip/hip_runtime.h>
#include <hip/hip_bf16.h>

typedef _Float16 v16h __attribute__((ext_vector_type(16)));
typedef _Float16 v8h  __attribute__((ext_vector_type(8)));
typedef float    v8f  __attribute__((ext_vector_type(8)));
typedef float    v4f  __attribute__((ext_vector_type(4)));

static __device__ __forceinline__ v8f wmma16(v16h a, v16h b, v8f c) {
    // D = A(16x32 f16) * B(32x16 f16) + C(16x16 f32)
    return __builtin_amdgcn_wmma_f32_16x16x32_f16(
        /*neg_a=*/false, a, /*neg_b=*/false, b,
        /*c_mod=*/(short)0, c, /*reuse_a=*/false, /*reuse_b=*/false);
}

// ---------------------------------------------------------------------------
// zero fill (float4 granularity)
// ---------------------------------------------------------------------------
__global__ void zero_kernel(v4f* __restrict__ p, int n4) {
    int i = blockIdx.x * blockDim.x + threadIdx.x;
    if (i < n4) { v4f z = {0.f, 0.f, 0.f, 0.f}; p[i] = z; }
}

// ---------------------------------------------------------------------------
// Pack an f32 weight matrix [nD][nK][64] into WMMA B-operand layout, f16,
// K zero-padded to nS*32.  Flat dst index = (((d*nS+s)*4+nt)*32+lane)*16+h
// B layout: K = 32s + 8*(lane>=16) + (h<8 ? h : h+8)  (h=0..15),
//           column N = nt*16 + (lane&15)
// ---------------------------------------------------------------------------
__global__ void pack_b_kernel(const float* __restrict__ src, _Float16* __restrict__ dst,
                              int nK, int nS, int total) {
    int idx = blockIdx.x * blockDim.x + threadIdx.x;
    if (idx >= total) return;
    int h    = idx & 15;
    int lane = (idx >> 4) & 31;
    int nt   = (idx >> 9) & 3;
    int rest = idx >> 11;
    int s    = rest % nS;
    int d    = rest / nS;
    int K = s * 32 + ((lane & 16) ? 8 : 0) + (h < 8 ? h : h + 8);
    int Nn = nt * 16 + (lane & 15);
    float v = (K < nK) ? src[((size_t)d * nK + K) * 64 + Nn] : 0.0f;
    dst[idx] = (_Float16)v;
}

// ---------------------------------------------------------------------------
// Edge gather + segment-sum: m_agg[dst, 0:64] += h[src]; m_agg[dst,64:80]+=ef
// m_agg is [N,96] f32, pad cols 80..95 stay zero.  32 lanes cooperate / edge.
// Atomics land in L2 (84MB state << 192MB L2).
// ---------------------------------------------------------------------------
template <bool F16SRC>
__global__ void scatter_kernel(const void* __restrict__ hsrc,
                               const float* __restrict__ edge_feat,
                               const int* __restrict__ esrc,
                               const int* __restrict__ edst,
                               const int* __restrict__ eeid,
                               float* __restrict__ m_agg, int nE) {
    long gid = (long)blockIdx.x * blockDim.x + threadIdx.x;
    int e    = (int)(gid >> 5);
    int lane = (int)(gid & 31);
    if (e >= nE) return;
    int s  = esrc[e];
    int d  = edst[e];
    int id = eeid[e];
    float v0, v1;
    if (F16SRC) {
        const _Float16* h = (const _Float16*)hsrc;
        v0 = (float)h[(size_t)s * 64 + lane];
        v1 = (float)h[(size_t)s * 64 + lane + 32];
    } else {
        const float* h = (const float*)hsrc;
        v0 = h[(size_t)s * 64 + lane];
        v1 = h[(size_t)s * 64 + lane + 32];
    }
    atomicAdd(&m_agg[(size_t)d * 96 + lane], v0);
    atomicAdd(&m_agg[(size_t)d * 96 + lane + 32], v1);
    if (lane < 16)
        atomicAdd(&m_agg[(size_t)d * 96 + 64 + lane], edge_feat[(size_t)id * 16 + lane]);
}

// ---------------------------------------------------------------------------
// Per-node update: out_all = m_agg @ U[d] for ALL 4 degree matrices (WMMA),
// then per-row select by deg_idx, sigmoid, store f16 h_out[N,64].
// One wave per 16-node tile; K = 96 (3 WMMA steps).
// Loop nest bounds live VGPRs: A cached (3x8), acc[4] (32) per column tile.
// ---------------------------------------------------------------------------
__global__ void update_kernel(const float* __restrict__ m_agg,
                              const _Float16* __restrict__ Bp,   // [4][3][4][32][16]
                              const int* __restrict__ deg_idx,
                              _Float16* __restrict__ h_out, int nTiles) {
    int wave = (int)((blockIdx.x * (long)blockDim.x + threadIdx.x) >> 5);
    int lane = threadIdx.x & 31;
    if (wave >= nTiles) return;
    int node0 = wave << 4;
    int row   = node0 + (lane & 15);
    int hi    = (lane & 16) ? 1 : 0;

    // load all three K-slices of A once (kept in 24 VGPRs)
    v16h A[3];
#pragma unroll
    for (int s = 0; s < 3; ++s) {
        int kbase = s * 32 + hi * 8;
        const v4f* pa = (const v4f*)(m_agg + (size_t)row * 96 + kbase);
        v4f a0 = pa[0], a1 = pa[1];   // K kbase   .. kbase+7
        v4f a2 = pa[4], a3 = pa[5];   // K kbase+16.. kbase+23
#pragma unroll
        for (int i = 0; i < 4; ++i) {
            A[s][i]      = (_Float16)a0[i];
            A[s][4 + i]  = (_Float16)a1[i];
            A[s][8 + i]  = (_Float16)a2[i];
            A[s][12 + i] = (_Float16)a3[i];
        }
    }

    // per-row degree selector (rows owned by this lane: M = v + hi*8)
    int dsel[8];
#pragma unroll
    for (int v = 0; v < 8; ++v) dsel[v] = deg_idx[node0 + v + hi * 8];

#pragma unroll
    for (int nt = 0; nt < 4; ++nt) {
        v8f acc[4] = {};  // one accumulator per degree matrix
#pragma unroll
        for (int s = 0; s < 3; ++s) {
#pragma unroll
            for (int d = 0; d < 4; ++d) {
                const v8h* bp = (const v8h*)(Bp + ((((d * 3 + s) * 4 + nt) * 32 + lane) << 4));
                v8h b0 = bp[0], b1 = bp[1];
                v16h B;
#pragma unroll
                for (int i = 0; i < 8; ++i) { B[i] = b0[i]; B[8 + i] = b1[i]; }
                acc[d] = wmma16(A[s], B, acc[d]);
            }
        }
#pragma unroll
        for (int v = 0; v < 8; ++v) {
            int node = node0 + v + hi * 8;
            int ds   = dsel[v];
            float x = (ds == 0) ? acc[0][v]
                    : (ds == 1) ? acc[1][v]
                    : (ds == 2) ? acc[2][v]
                                : acc[3][v];
            float y = 1.0f / (1.0f + __expf(-x));
            h_out[(size_t)node * 64 + nt * 16 + (lane & 15)] = (_Float16)y;
        }
    }
}

// ---------------------------------------------------------------------------
// Readout: S = h_tile(16x64) @ R(64x64) via WMMA; row-softmax in registers
// (shuffle reductions across the 16-lane half that owns each row), column
// sums -> LDS fp[64] -> global atomic fp[64].
// ---------------------------------------------------------------------------
template <bool F16SRC>
__global__ void readout_kernel(const void* __restrict__ hsrc,
                               const _Float16* __restrict__ Rp,  // [2][4][32][16]
                               float* __restrict__ fp, int nTiles) {
    __shared__ float fpl[64];
    if (threadIdx.x < 64) fpl[threadIdx.x] = 0.0f;
    __syncthreads();

    int wave = (int)((blockIdx.x * (long)blockDim.x + threadIdx.x) >> 5);
    int lane = threadIdx.x & 31;
    if (wave < nTiles) {
        int node0 = wave << 4;
        int row   = node0 + (lane & 15);
        int hi    = (lane & 16) ? 1 : 0;
        v8f acc[4] = {};

#pragma unroll
        for (int s = 0; s < 2; ++s) {
            int kbase = s * 32 + hi * 8;
            v16h A;
            if (F16SRC) {
                const _Float16* h = (const _Float16*)hsrc;
                const v8h* pa = (const v8h*)(h + (size_t)row * 64 + kbase);
                v8h a0 = pa[0], a1 = pa[2];
#pragma unroll
                for (int i = 0; i < 8; ++i) { A[i] = a0[i]; A[8 + i] = a1[i]; }
            } else {
                const float* h = (const float*)hsrc;
                const v4f* pa = (const v4f*)(h + (size_t)row * 64 + kbase);
                v4f a0 = pa[0], a1 = pa[1], a2 = pa[4], a3 = pa[5];
#pragma unroll
                for (int i = 0; i < 4; ++i) {
                    A[i]      = (_Float16)a0[i];
                    A[4 + i]  = (_Float16)a1[i];
                    A[8 + i]  = (_Float16)a2[i];
                    A[12 + i] = (_Float16)a3[i];
                }
            }
#pragma unroll
            for (int nt = 0; nt < 4; ++nt) {
                const v8h* bp = (const v8h*)(Rp + (((s * 4 + nt) * 32 + lane) << 4));
                v8h b0 = bp[0], b1 = bp[1];
                v16h B;
#pragma unroll
                for (int i = 0; i < 8; ++i) { B[i] = b0[i]; B[8 + i] = b1[i]; }
                acc[nt] = wmma16(A, B, acc[nt]);
            }
        }

        float colsum[4] = {0.f, 0.f, 0.f, 0.f};
#pragma unroll
        for (int v = 0; v < 8; ++v) {
            float mx = fmaxf(fmaxf(acc[0][v], acc[1][v]), fmaxf(acc[2][v], acc[3][v]));
#pragma unroll
            for (int off = 1; off < 16; off <<= 1)
                mx = fmaxf(mx, __shfl_xor(mx, off, 32));
            float e0 = __expf(acc[0][v] - mx);
            float e1 = __expf(acc[1][v] - mx);
            float e2 = __expf(acc[2][v] - mx);
            float e3 = __expf(acc[3][v] - mx);
            float sum = e0 + e1 + e2 + e3;
#pragma unroll
            for (int off = 1; off < 16; off <<= 1)
                sum += __shfl_xor(sum, off, 32);
            float inv = 1.0f / sum;
            colsum[0] += e0 * inv;
            colsum[1] += e1 * inv;
            colsum[2] += e2 * inv;
            colsum[3] += e3 * inv;
        }
#pragma unroll
        for (int nt = 0; nt < 4; ++nt)
            colsum[nt] += __shfl_xor(colsum[nt], 16, 32);
        if (lane < 16) {
#pragma unroll
            for (int nt = 0; nt < 4; ++nt)
                atomicAdd(&fpl[nt * 16 + lane], colsum[nt]);
        }
    }
    __syncthreads();
    if (threadIdx.x < 64) atomicAdd(&fp[threadIdx.x], fpl[threadIdx.x]);
}

// ---------------------------------------------------------------------------
// out[t] = fp @ W_out[:,t] + b_out[t]  (64x12, one lane per target)
// ---------------------------------------------------------------------------
__global__ void final_kernel(const float* __restrict__ fp,
                             const float* __restrict__ W_out,
                             const float* __restrict__ b_out,
                             float* __restrict__ out) {
    int t = threadIdx.x;
    if (t < 12) {
        float s = b_out[t];
#pragma unroll
        for (int c = 0; c < 64; ++c) s += fp[c] * W_out[c * 12 + t];
        out[t] = s;
    }
}

// ---------------------------------------------------------------------------
extern "C" void kernel_launch(void* const* d_in, const int* in_sizes, int n_in,
                              void* d_out, int out_size, void* d_ws, size_t ws_size,
                              hipStream_t stream) {
    const float* h_in      = (const float*)d_in[0];
    const float* edge_feat = (const float*)d_in[1];
    const float* U0        = (const float*)d_in[2];
    const float* U1        = (const float*)d_in[3];
    const float* R0        = (const float*)d_in[4];
    const float* R1        = (const float*)d_in[5];
    const float* R2        = (const float*)d_in[6];
    const float* W_out     = (const float*)d_in[7];
    const float* b_out     = (const float*)d_in[8];
    const int*   esrc      = (const int*)d_in[9];
    const int*   edst      = (const int*)d_in[10];
    const int*   eeid      = (const int*)d_in[11];
    const int*   deg_idx   = (const int*)d_in[12];

    const int N      = in_sizes[0] / 64;   // 262144
    const int twoE   = in_sizes[9];        // directed edges
    const int nTiles = N / 16;

    // workspace carve-up (256B aligned)
    char* ws = (char*)d_ws;
    size_t off = 0;
    auto carve = [&](size_t bytes) -> char* {
        char* p = ws + off;
        off = (off + bytes + 255) & ~(size_t)255;
        return p;
    };
    float*     m_agg = (float*)    carve((size_t)N * 96 * sizeof(float));      // ~100MB, fits L2
    _Float16*  h1    = (_Float16*) carve((size_t)N * 64 * sizeof(_Float16));   // 32MB
    _Float16*  h2    = (_Float16*) carve((size_t)N * 64 * sizeof(_Float16));   // 32MB
    _Float16*  BU0   = (_Float16*) carve((size_t)4 * 3 * 4 * 32 * 16 * 2);
    _Float16*  BU1   = (_Float16*) carve((size_t)4 * 3 * 4 * 32 * 16 * 2);
    _Float16*  BR0   = (_Float16*) carve((size_t)2 * 4 * 32 * 16 * 2);
    _Float16*  BR1   = (_Float16*) carve((size_t)2 * 4 * 32 * 16 * 2);
    _Float16*  BR2   = (_Float16*) carve((size_t)2 * 4 * 32 * 16 * 2);
    float*     fp    = (float*)    carve(64 * sizeof(float));
    (void)ws_size;

    const int TPB = 256;

    // pack weights into WMMA B layout (f16, K padded)
    const int totU = 4 * 3 * 4 * 32 * 16;
    const int totR = 1 * 2 * 4 * 32 * 16;
    pack_b_kernel<<<(totU + TPB - 1) / TPB, TPB, 0, stream>>>(U0, BU0, 80, 3, totU);
    pack_b_kernel<<<(totU + TPB - 1) / TPB, TPB, 0, stream>>>(U1, BU1, 80, 3, totU);
    pack_b_kernel<<<(totR + TPB - 1) / TPB, TPB, 0, stream>>>(R0, BR0, 64, 2, totR);
    pack_b_kernel<<<(totR + TPB - 1) / TPB, TPB, 0, stream>>>(R1, BR1, 64, 2, totR);
    pack_b_kernel<<<(totR + TPB - 1) / TPB, TPB, 0, stream>>>(R2, BR2, 64, 2, totR);

    const int magg4 = N * 96 / 4;
    const long scatterThreads = (long)twoE * 32;
    const int  scatterBlocks  = (int)((scatterThreads + TPB - 1) / TPB);
    const int  tileBlocks     = (nTiles * 32 + TPB - 1) / TPB;

    // layer 0
    zero_kernel<<<(magg4 + TPB - 1) / TPB, TPB, 0, stream>>>((v4f*)m_agg, magg4);
    scatter_kernel<false><<<scatterBlocks, TPB, 0, stream>>>(h_in, edge_feat, esrc, edst, eeid, m_agg, twoE);
    update_kernel<<<tileBlocks, TPB, 0, stream>>>(m_agg, BU0, deg_idx, h1, nTiles);

    // layer 1
    zero_kernel<<<(magg4 + TPB - 1) / TPB, TPB, 0, stream>>>((v4f*)m_agg, magg4);
    scatter_kernel<true><<<scatterBlocks, TPB, 0, stream>>>(h1, edge_feat, esrc, edst, eeid, m_agg, twoE);
    update_kernel<<<tileBlocks, TPB, 0, stream>>>(m_agg, BU1, deg_idx, h2, nTiles);

    // readout fingerprint
    zero_kernel<<<1, 64, 0, stream>>>((v4f*)fp, 16);
    readout_kernel<false><<<tileBlocks, TPB, 0, stream>>>(h_in, BR0, fp, nTiles);
    readout_kernel<true ><<<tileBlocks, TPB, 0, stream>>>(h1,   BR1, fp, nTiles);
    readout_kernel<true ><<<tileBlocks, TPB, 0, stream>>>(h2,   BR2, fp, nTiles);

    // final linear
    final_kernel<<<1, 64, 0, stream>>>(fp, W_out, b_out, (float*)d_out);
}
